// Net_29643864277324
// MI455X (gfx1250) — matrix operand order repaired and checked by reading, not compiled, not run on hardware
//
#include <hip/hip_runtime.h>
#include <hip/hip_bf16.h>
#include <math.h>

typedef __attribute__((ext_vector_type(16))) __bf16 v16bf;
typedef __attribute__((ext_vector_type(8)))  float  v8f;

#define N_NODES    50000
#define N_EDGES    600000
#define DIM        128
#define NUM_GRAPHS 512

// packed weight fragments: 2 matrices * 32 frags * 32 lanes * 16 bf16 = 32768 bf16 (64 KB)
#define NFRAG      32            // 8 n-tiles * 4 k-chunks per matrix
#define WPACK_ELEMS (2 * NFRAG * 32 * 16)

// ---------------- zero scratch ----------------
__global__ void k_zero(float* __restrict__ p, int n) {
  int i = blockIdx.x * blockDim.x + threadIdx.x;
  int stride = gridDim.x * blockDim.x;
  for (; i < n; i += stride) p[i] = 0.0f;
}

// ---------------- pack W_l / W_r into bf16 WMMA B-fragment layout ----------------
// frag index f = (m*NFRAG + nt*4 + kc); within frag: lane holds col N = nt*16 + lane%16,
// element e -> K = kc*32 + (e>>3)*16 + (lane>>4)*8 + (e&7); B[k][n] = W[n][k] (B = W^T).
__global__ void k_pack_w(const float* __restrict__ Wl, const float* __restrict__ Wr,
                         __bf16* __restrict__ wpack) {
  int i = blockIdx.x * blockDim.x + threadIdx.x;
  if (i >= WPACK_ELEMS) return;
  int e    = i & 15;
  int lane = (i >> 4) & 31;
  int kc   = (i >> 9) & 3;
  int nt   = (i >> 11) & 7;
  int m    = (i >> 14) & 1;
  int n = nt * 16 + (lane & 15);
  int k = kc * 32 + (e >> 3) * 16 + (lane >> 4) * 8 + (e & 7);
  const float* W = m ? Wr : Wl;
  wpack[i] = (__bf16)W[(size_t)n * DIM + k];
}

// ---------------- embedding gather: x[n] = emb[ids[n]] ----------------
__global__ void k_gather(const int* __restrict__ ids, const float* __restrict__ emb,
                         float* __restrict__ x) {
  int i = blockIdx.x * blockDim.x + threadIdx.x;     // one float4 per thread
  if (i >= N_NODES * (DIM / 4)) return;
  int n = i >> 5;                                    // DIM/4 == 32 chunks per node
  int c = i & 31;
  const float4* s = (const float4*)(emb + (size_t)ids[n] * DIM);
  ((float4*)(x + (size_t)n * DIM))[c] = s[c];
}

// ---------------- edge scatter: agg[dst] += x[src], deg[dst] += 1 ----------------
__global__ void k_scatter(const int* __restrict__ src, const int* __restrict__ dst,
                          const float* __restrict__ x, float* __restrict__ agg,
                          float* __restrict__ deg) {
  int e = blockIdx.x * (blockDim.x >> 5) + (threadIdx.x >> 5);   // one wave32 per edge
  int lane = threadIdx.x & 31;
  if (e >= N_EDGES) return;
  int s = src[e], d = dst[e];
  const float* xr = x + (size_t)s * DIM;
  float* ar = agg + (size_t)d * DIM;
#pragma unroll
  for (int i = 0; i < 4; ++i) {
    int c = lane + i * 32;
    atomicAdd(&ar[c], xr[c]);
  }
  if (lane == 0) atomicAdd(&deg[d], 1.0f);
}

// ------- h = relu((agg/deg) @ Wl^T + bl + x @ Wr^T), written in place over agg -------
// Weights staged once per block into LDS as prepacked bf16 fragments; one wave32 per
// 16-row tile; 8 N-tiles x (4 K-chunks x 2 matrices) = 64 WMMAs/wave.
__global__ void k_gemm_relu(const float* __restrict__ x, float* __restrict__ aggh,
                            const float* __restrict__ deg,
                            const __bf16* __restrict__ wpack,
                            const float* __restrict__ bl) {
  __shared__ v16bf lds_w[2 * NFRAG * 32];            // 64 KB

  // cooperative global -> LDS copy (4096 uint4 across 256 threads)
  {
    const uint4* gsrc = (const uint4*)wpack;
    uint4* ldst = (uint4*)lds_w;
    for (int i = threadIdx.x; i < (WPACK_ELEMS * 2) / 16; i += 256) ldst[i] = gsrc[i];
  }
  __syncthreads();

  const int wave = threadIdx.x >> 5;
  const int lane = threadIdx.x & 31;
  const int tile = blockIdx.x * 8 + wave;
  if (tile * 16 >= N_NODES) return;                  // wave-uniform: EXEC stays all-ones
  const int row0 = tile * 16;
  const int mrow = lane & 15;                        // A-matrix: M = lane % 16
  const int hi   = lane >> 4;                        // upper half-wave -> K offset +8
  const int r    = row0 + mrow;                      // 50000 % 16 == 0: always valid

  const float ds = 1.0f / fmaxf(deg[r], 1.0f);
  const float* aggRow = aggh + (size_t)r * DIM;
  const float* xRow   = x    + (size_t)r * DIM;

  // A operands per 16-bit 16x32 layout: elems 0..7 -> K = k0..k0+7, 8..15 -> K = k0+16..k0+23
  v16bf a_agg[4], a_x[4];
#pragma unroll
  for (int kc = 0; kc < 4; ++kc) {
    const int k0 = kc * 32 + hi * 8;
#pragma unroll
    for (int e = 0; e < 8; ++e) {
      a_agg[kc][e]     = (__bf16)(aggRow[k0 + e]      * ds);
      a_agg[kc][e + 8] = (__bf16)(aggRow[k0 + 16 + e] * ds);
      a_x[kc][e]       = (__bf16)(xRow[k0 + e]);
      a_x[kc][e + 8]   = (__bf16)(xRow[k0 + 16 + e]);
    }
  }

  const int ncl = lane & 15;                         // B/C/D: N = lane % 16
#pragma unroll
  for (int nt = 0; nt < 8; ++nt) {
    const int n = nt * 16 + ncl;
    const float bias = bl[n];
    v8f c = {bias, bias, bias, bias, bias, bias, bias, bias};
#pragma unroll
    for (int kc = 0; kc < 4; ++kc) {
      v16bf bwl = lds_w[(nt * 4 + kc) * 32 + lane];                  // W_l^T fragment
      v16bf bwr = lds_w[NFRAG * 32 + (nt * 4 + kc) * 32 + lane];     // W_r^T fragment
      c = __builtin_amdgcn_wmma_f32_16x16x32_bf16(false, a_agg[kc], false, bwl,
                                                  (short)0, c, false, false);
      c = __builtin_amdgcn_wmma_f32_16x16x32_bf16(false, a_x[kc],   false, bwr,
                                                  (short)0, c, false, false);
    }
    // D layout: vgpr rr -> row rr + 8*hi, col = n
#pragma unroll
    for (int rr = 0; rr < 8; ++rr) {
      float v = fmaxf(c[rr], 0.0f);
      aggh[(size_t)(row0 + rr + hi * 8) * DIM + n] = v;
    }
  }
}

// ---------------- per-graph pooling: max (bit-trick, h>=0) and sum + count ----------------
__global__ void k_pool(const float* __restrict__ h, const int* __restrict__ batch,
                       float* __restrict__ hmax, float* __restrict__ hsum,
                       float* __restrict__ cnt) {
  int n = blockIdx.x * (blockDim.x >> 5) + (threadIdx.x >> 5);   // one wave32 per node
  int lane = threadIdx.x & 31;
  if (n >= N_NODES) return;
  int g = batch[n];
  const float* hr = h + (size_t)n * DIM;
  float* mx = hmax + (size_t)g * DIM;
  float* sm = hsum + (size_t)g * DIM;
#pragma unroll
  for (int i = 0; i < 4; ++i) {
    int c = lane + i * 32;
    float v = hr[c];
    atomicAdd(&sm[c], v);
    atomicMax((int*)&mx[c], __float_as_int(v));      // valid: v >= 0, init 0
  }
  if (lane == 0) atomicAdd(&cnt[g], 1.0f);
}

// ---------------- out[g] = sigmoid([hmax|hmean] . lin1_W + lin1_b) ----------------
__global__ void k_final(const float* __restrict__ hmax, const float* __restrict__ hsum,
                        const float* __restrict__ cnt, const float* __restrict__ W,
                        const float* __restrict__ b, float* __restrict__ out) {
  int g = blockIdx.x * (blockDim.x >> 5) + (threadIdx.x >> 5);   // one wave32 per graph
  int lane = threadIdx.x & 31;
  if (g >= NUM_GRAPHS) return;
  float inv = 1.0f / fmaxf(cnt[g], 1.0f);
  float acc = 0.0f;
#pragma unroll
  for (int i = 0; i < 4; ++i) {
    int c = lane + i * 32;
    acc += hmax[(size_t)g * DIM + c] * W[c];
    acc += (hsum[(size_t)g * DIM + c] * inv) * W[DIM + c];
  }
#pragma unroll
  for (int off = 16; off > 0; off >>= 1) acc += __shfl_down(acc, off, 32);
  if (lane == 0) out[g] = 1.0f / (1.0f + expf(-(acc + b[0])));
}

extern "C" void kernel_launch(void* const* d_in, const int* in_sizes, int n_in,
                              void* d_out, int out_size, void* d_ws, size_t ws_size,
                              hipStream_t stream) {
  const int*   x_ids  = (const int*)d_in[0];
  const int*   eidx   = (const int*)d_in[1];      // (2, N_EDGES)
  const int*   batch  = (const int*)d_in[2];
  const float* emb    = (const float*)d_in[3];
  const float* W_l    = (const float*)d_in[4];
  const float* b_l    = (const float*)d_in[5];
  const float* W_r    = (const float*)d_in[6];
  const float* lin1_W = (const float*)d_in[7];
  const float* lin1_b = (const float*)d_in[8];
  float* out = (float*)d_out;

  const int* src = eidx;
  const int* dst = eidx + N_EDGES;

  // workspace layout (floats; every segment is a multiple of 4 floats -> 16 B aligned)
  float* ws   = (float*)d_ws;
  float* x    = ws;                                  // N_NODES*DIM
  float* aggh = x + (size_t)N_NODES * DIM;           // N_NODES*DIM (agg -> h in place)
  float* deg  = aggh + (size_t)N_NODES * DIM;        // N_NODES
  float* hmax = deg + N_NODES;                       // NUM_GRAPHS*DIM
  float* hsum = hmax + (size_t)NUM_GRAPHS * DIM;     // NUM_GRAPHS*DIM
  float* cnt  = hsum + (size_t)NUM_GRAPHS * DIM;     // NUM_GRAPHS
  __bf16* wpack = (__bf16*)(cnt + NUM_GRAPHS);       // WPACK_ELEMS bf16 (64 KB)

  const int zero_n = (int)((size_t)N_NODES * DIM + N_NODES +
                           2 * (size_t)NUM_GRAPHS * DIM + NUM_GRAPHS);
  k_zero<<<2048, 256, 0, stream>>>(aggh, zero_n);

  k_pack_w<<<(WPACK_ELEMS + 255) / 256, 256, 0, stream>>>(W_l, W_r, wpack);

  k_gather<<<(N_NODES * (DIM / 4) + 255) / 256, 256, 0, stream>>>(x_ids, emb, x);

  k_scatter<<<(N_EDGES + 7) / 8, 256, 0, stream>>>(src, dst, x, aggh, deg);

  const int tiles = N_NODES / 16;                    // 3125, exact
  k_gemm_relu<<<(tiles + 7) / 8, 256, 0, stream>>>(x, aggh, deg, wpack, b_l);

  k_pool<<<(N_NODES + 7) / 8, 256, 0, stream>>>(aggh, batch, hmax, hsum, cnt);

  k_final<<<(NUM_GRAPHS + 7) / 8, 256, 0, stream>>>(hmax, hsum, cnt, lin1_W, lin1_b, out);
}
